// AxialMultiHeadMixAttention_3126736192155
// MI455X (gfx1250) — matrix operand order repaired and checked by reading, not compiled, not run on hardware
//
#include <hip/hip_runtime.h>

// Problem constants (match reference)
#define BB    8
#define LL    128
#define TT    32
#define DD    128
#define HH    8
#define DKK   16
#define SEEDL 64
#define TGTL  64
#define SCALE 0.25f      // 1/sqrt(DK)
#define NEGV  -1e9f

typedef __attribute__((ext_vector_type(16))) __bf16        v16bf;
typedef __attribute__((ext_vector_type(8)))  float         v8f;
typedef __attribute__((ext_vector_type(4)))  unsigned int  u32x4;   // 16B aligned
typedef __attribute__((ext_vector_type(4)))  float         f32x4;   // 16B aligned

__device__ __forceinline__ unsigned short f2bf(float f) {
  unsigned int u = __float_as_uint(f);
  u += 0x7FFFu + ((u >> 16) & 1u);           // round-to-nearest-even
  return (unsigned short)(u >> 16);
}

union FragBF { v16bf v; u32x4 q[2]; };
union Pack8  { u32x4 q; unsigned short h[8]; };

__device__ __forceinline__ v8f vzero() {
  v8f z;
#pragma unroll
  for (int i = 0; i < 8; ++i) z[i] = 0.0f;
  return z;
}

__device__ __forceinline__ v8f wmma_bf(v16bf a, v16bf b, v8f c) {
  return __builtin_amdgcn_wmma_f32_16x16x32_bf16(false, a, false, b, (short)0, c, false, false);
}

// A fragment: 16x32 bf16 tile from LDS row-major [M][K], stride ld (ushort units).
// Lane layout (ISA 7.12.2): M = lane%16; lanes 0-15 hold K {0..7,16..23}, lanes 16-31 K {8..15,24..31}.
// Both halves are 16B-aligned -> two ds_load_b128.
__device__ __forceinline__ v16bf ldsA(const unsigned short* sh, int ld, int r0, int c0, int lane) {
  const int m = lane & 15, g = lane >> 4;
  const unsigned short* p = sh + (r0 + m) * ld + c0 + g * 8;
  FragBF f;
  f.q[0] = *(const u32x4*)p;
  f.q[1] = *(const u32x4*)(p + 16);
  return f.v;
}

// A fragment with only K=0..15 valid (upper 16 K zero) — for DK=16 score matmuls.
__device__ __forceinline__ v16bf ldsA16(const unsigned short* sh, int ld, int r0, int c0, int lane) {
  const int m = lane & 15, g = lane >> 4;
  FragBF f;
  f.q[0] = *(const u32x4*)(sh + (r0 + m) * ld + c0 + g * 8);
  f.q[1] = (u32x4){0u, 0u, 0u, 0u};
  return f.v;
}

// B fragment: 32x16 bf16 (KxN) from LDS laid out [N][K] (K contiguous per n), stride ld.
// Lanes 0-15 hold K {0..7,16..23}, lanes 16-31 K {8..15,24..31}; N = lane%16.
__device__ __forceinline__ v16bf ldsB(const unsigned short* sh, int ld, int n0, int k0, int lane) {
  const int n = lane & 15, g = lane >> 4;
  const unsigned short* p = sh + (n0 + n) * ld + k0 + g * 8;
  FragBF f;
  f.q[0] = *(const u32x4*)p;
  f.q[1] = *(const u32x4*)(p + 16);
  return f.v;
}

// B fragment with only K=0..15 valid.
__device__ __forceinline__ v16bf ldsB16(const unsigned short* sh, int ld, int n0, int k0, int lane) {
  const int n = lane & 15, g = lane >> 4;
  const unsigned short* p = sh + (n0 + n) * ld + k0 + g * 8;
  FragBF f;
  u32x4 a0 = *(const u32x4*)p;
  u32x4 z  = (u32x4){0u, 0u, 0u, 0u};
  f.q[0] = a0;
  f.q[1] = z;
  return f.v;
}

// Store f32 C tile (16x16) to bf16 LDS row-major, or transposed.
__device__ __forceinline__ void stC(unsigned short* sh, int ld, int r0, int c0, int lane, v8f a) {
  const int n = lane & 15, g = lane >> 4;
#pragma unroll
  for (int r = 0; r < 8; ++r) sh[(r0 + g * 8 + r) * ld + c0 + n] = f2bf(a[r]);
}
__device__ __forceinline__ void stCT(unsigned short* sh, int ld, int r0, int c0, int lane, v8f a) {
  const int n = lane & 15, g = lane >> 4;
#pragma unroll
  for (int r = 0; r < 8; ++r) sh[(c0 + n) * ld + r0 + g * 8 + r] = f2bf(a[r]);
}

// Convert 8 consecutive floats to one 16B bf16 vector.
__device__ __forceinline__ u32x4 cvt8(const float* src) {
  f32x4 a = *(const f32x4*)src;
  f32x4 b = *(const f32x4*)(src + 4);
  Pack8 o;
#pragma unroll
  for (int j = 0; j < 4; ++j) { o.h[j] = f2bf(a[j]); o.h[4 + j] = f2bf(b[j]); }
  return o.q;
}

// ---------------------------------------------------------------------------
// Kernel 0: convert the 8 DxD weight matrices to bf16 in workspace (vectorized).
// ---------------------------------------------------------------------------
__global__ void prep_weights_kernel(const float* w0, const float* w1, const float* w2,
                                    const float* w3, const float* w4, const float* w5,
                                    const float* w6, const float* w7,
                                    unsigned short* dst) {
  int i = blockIdx.x * blockDim.x + threadIdx.x;   // 8*16384/8 = 16384 vectors
  int m = i >> 11, j = i & 2047;                   // 2048 vecs per matrix
  const float* srcs[8] = {w0, w1, w2, w3, w4, w5, w6, w7};
  ((u32x4*)dst)[i] = cvt8(srcs[m] + j * 8);
}

// ---------------------------------------------------------------------------
// Kernel 1: temporal stage. One workgroup per (b,t); 8 waves. 224 KB LDS.
// ---------------------------------------------------------------------------
__global__ void __launch_bounds__(256)
temporal_kernel(const float* __restrict__ qg, const float* __restrict__ kg,
                const float* __restrict__ vg, const unsigned char* __restrict__ mask,
                const unsigned short* __restrict__ wq, const unsigned short* __restrict__ wk,
                const unsigned short* __restrict__ wv, const unsigned short* __restrict__ wd,
                const float* __restrict__ bd, unsigned short* __restrict__ xt) {
  extern __shared__ unsigned short sm[];
  unsigned short* sA  = sm;                 // input (bf16)      [128][128]
  unsigned short* sW  = sm + 1 * 16384;     // weight (bf16)     [c][k]
  unsigned short* sQ  = sm + 2 * 16384;     // qt                [l][d]
  unsigned short* sK  = sm + 3 * 16384;     // kt                [l][d]
  unsigned short* sVt = sm + 4 * 16384;     // vt transposed     [d][l]
  unsigned short* sO  = sm + 5 * 16384;     // attention output  [l][d]
  unsigned short* sP  = sm + 6 * 16384;     // per-wave P        [16][128]

  const int b = blockIdx.x / TT, t = blockIdx.x % TT;
  const int tid = threadIdx.x, w = tid >> 5, lane = tid & 31;
  const size_t xbase = ((size_t)b * LL) * (TT * DD) + (size_t)t * DD;

  // ---- projections qt/kt/vt = X @ W.T ------------------------------------
  const float* xs[3] = {qg, kg, vg};
  const unsigned short* wsrc[3] = {wq, wk, wv};
#pragma unroll
  for (int p = 0; p < 3; ++p) {
    {
      u32x4* sAv = (u32x4*)sA;
      u32x4* sWv = (u32x4*)sW;
      const u32x4* wv4 = (const u32x4*)wsrc[p];
      const float* xsrc = xs[p];
#pragma unroll
      for (int it = 0; it < 8; ++it) {            // 2048 vecs / 256 threads
        int i = tid + it * 256;
        int l = i >> 4, d = (i & 15) * 8;         // 16 vecs per 128-elem row
        sAv[i] = cvt8(xsrc + xbase + (size_t)l * (TT * DD) + d);
        sWv[i] = wv4[i];
      }
    }
    __syncthreads();
    v8f acc[8];
#pragma unroll
    for (int nt = 0; nt < 8; ++nt) acc[nt] = vzero();
#pragma unroll
    for (int kk = 0; kk < 4; ++kk) {
      v16bf a = ldsA(sA, 128, w * 16, kk * 32, lane);
#pragma unroll
      for (int nt = 0; nt < 8; ++nt)
        acc[nt] = wmma_bf(a, ldsB(sW, 128, nt * 16, kk * 32, lane), acc[nt]);
    }
#pragma unroll
    for (int nt = 0; nt < 8; ++nt) {
      if (p == 0)      stC (sQ,  128, w * 16, nt * 16, lane, acc[nt]);
      else if (p == 1) stC (sK,  128, w * 16, nt * 16, lane, acc[nt]);
      else             stCT(sVt, 128, w * 16, nt * 16, lane, acc[nt]);
    }
    __syncthreads();
  }

  // ---- attention: one wave per head --------------------------------------
  const int h = w;
  unsigned short* Pw = sP + w * (16 * 128);
  const unsigned char* mbase = mask + (size_t)(b * TT + t) * TGTL * LL;
  const int n = lane & 15, g = lane >> 4;

  for (int mt = 0; mt < 8; ++mt) {
    const int q0 = mt * 16;
    const bool seed = mt < 4;                 // seed queries attend keys [0,64)
    const int nkt = seed ? 4 : 8;             // key tiles
    v16bf af = ldsA16(sQ, 128, q0, h * 16, lane);
    v8f sc[8];
#pragma unroll
    for (int nt = 0; nt < 8; ++nt) sc[nt] = vzero();
    for (int nt = 0; nt < nkt; ++nt)
      sc[nt] = wmma_bf(af, ldsB16(sK, 128, nt * 16, h * 16, lane), sc[nt]);

    // scale + mask (target queries only)
    for (int nt = 0; nt < nkt; ++nt) {
#pragma unroll
      for (int r = 0; r < 8; ++r) {
        float s = sc[nt][r] * SCALE;
        if (!seed) {
          int qt_ = q0 - SEEDL + g * 8 + r;
          int key = nt * 16 + n;
          if (!mbase[qt_ * LL + key]) s = NEGV;
        }
        sc[nt][r] = s;
      }
    }
    // softmax per row (each row lives across one 16-lane half)
#pragma unroll
    for (int r = 0; r < 8; ++r) {
      float mx = -3.0e38f;
      for (int nt = 0; nt < nkt; ++nt) mx = fmaxf(mx, sc[nt][r]);
#pragma unroll
      for (int off = 1; off < 16; off <<= 1) mx = fmaxf(mx, __shfl_xor(mx, off, 32));
      float sum = 0.0f;
      for (int nt = 0; nt < nkt; ++nt) { float e = __expf(sc[nt][r] - mx); sc[nt][r] = e; sum += e; }
#pragma unroll
      for (int off = 1; off < 16; off <<= 1) sum += __shfl_xor(sum, off, 32);
      float inv = 1.0f / sum;
      for (int nt = 0; nt < nkt; ++nt)
        Pw[(g * 8 + r) * 128 + nt * 16 + n] = f2bf(sc[nt][r] * inv);
    }
    // O = P @ V_h  (V stored transposed so B fragments are contiguous)
    v8f o = vzero();
    const int nkc = seed ? 2 : 4;
    for (int kc = 0; kc < nkc; ++kc)
      o = wmma_bf(ldsA(Pw, 128, 0, kc * 32, lane), ldsB(sVt, 128, h * 16, kc * 32, lane), o);
#pragma unroll
    for (int r = 0; r < 8; ++r)
      sO[(q0 + g * 8 + r) * 128 + h * 16 + n] = f2bf(o[r]);
  }

  // ---- output projection + bias, store bf16 to x_t[b][l][t][d] -----------
  {
    u32x4* sWv = (u32x4*)sW;
    const u32x4* wv4 = (const u32x4*)wd;
#pragma unroll
    for (int it = 0; it < 8; ++it) sWv[tid + it * 256] = wv4[tid + it * 256];
  }
  __syncthreads();
  v8f acc[8];
#pragma unroll
  for (int nt = 0; nt < 8; ++nt) acc[nt] = vzero();
#pragma unroll
  for (int kk = 0; kk < 4; ++kk) {
    v16bf a = ldsA(sO, 128, w * 16, kk * 32, lane);
#pragma unroll
    for (int nt = 0; nt < 8; ++nt)
      acc[nt] = wmma_bf(a, ldsB(sW, 128, nt * 16, kk * 32, lane), acc[nt]);
  }
#pragma unroll
  for (int nt = 0; nt < 8; ++nt) {
#pragma unroll
    for (int r = 0; r < 8; ++r) {
      int l = w * 16 + g * 8 + r, d = nt * 16 + n;
      xt[xbase + (size_t)l * (TT * DD) + d] = f2bf(acc[nt][r] + bd[d]);
    }
  }
}

// ---------------------------------------------------------------------------
// Kernel 2: social stage. One workgroup per (b,l); 8 waves. ~80 KB LDS.
// ---------------------------------------------------------------------------
__global__ void __launch_bounds__(256)
social_kernel(const unsigned short* __restrict__ xt,
              const unsigned short* __restrict__ wq, const unsigned short* __restrict__ wk,
              const unsigned short* __restrict__ wv, const unsigned short* __restrict__ wd,
              const float* __restrict__ bd, float* __restrict__ out) {
  extern __shared__ unsigned short sm[];
  unsigned short* sX  = sm;            // [32][128]
  unsigned short* sQ  = sm + 4096;     // [32][128]
  unsigned short* sK  = sm + 8192;     // [32][128]
  unsigned short* sVt = sm + 12288;    // [128][32]
  unsigned short* sO  = sm + 16384;    // [32][128]
  unsigned short* sP  = sm + 20480;    // 8 x [16][32]
  unsigned short* sW  = sm + 24576;    // [128][128]

  const int tid = threadIdx.x, w = tid >> 5, lane = tid & 31;
  const size_t base = (size_t)blockIdx.x * (TT * DD);
  const int n = lane & 15, g = lane >> 4;

  {
    u32x4* sXv = (u32x4*)sX;
    const u32x4* xv = (const u32x4*)(xt + base);
#pragma unroll
    for (int it = 0; it < 2; ++it) sXv[tid + it * 256] = xv[tid + it * 256];  // 512 vecs
  }

  const unsigned short* wsrc[3] = {wq, wk, wv};
  const int mt = w >> 2, ntb = (w & 3) * 2;   // wave tile assignment for 2x8 tile grid
#pragma unroll
  for (int p = 0; p < 3; ++p) {
    {
      u32x4* sWv = (u32x4*)sW;
      const u32x4* wv4 = (const u32x4*)wsrc[p];
#pragma unroll
      for (int it = 0; it < 8; ++it) sWv[tid + it * 256] = wv4[tid + it * 256];
    }
    __syncthreads();
    v8f acc[2] = {vzero(), vzero()};
#pragma unroll
    for (int kk = 0; kk < 4; ++kk) {
      v16bf a = ldsA(sX, 128, mt * 16, kk * 32, lane);
#pragma unroll
      for (int j = 0; j < 2; ++j)
        acc[j] = wmma_bf(a, ldsB(sW, 128, (ntb + j) * 16, kk * 32, lane), acc[j]);
    }
#pragma unroll
    for (int j = 0; j < 2; ++j) {
      if (p == 0)      stC (sQ,  128, mt * 16, (ntb + j) * 16, lane, acc[j]);
      else if (p == 1) stC (sK,  128, mt * 16, (ntb + j) * 16, lane, acc[j]);
      else             stCT(sVt, 32,  mt * 16, (ntb + j) * 16, lane, acc[j]);
    }
    __syncthreads();
  }

  // attention over T=32 tracks, one wave per head
  const int h = w;
  unsigned short* Pw = sP + w * (16 * 32);
#pragma unroll
  for (int qt_ = 0; qt_ < 2; ++qt_) {
    const int q0 = qt_ * 16;
    v16bf af = ldsA16(sQ, 128, q0, h * 16, lane);
    v8f sc[2] = {vzero(), vzero()};
#pragma unroll
    for (int nt = 0; nt < 2; ++nt)
      sc[nt] = wmma_bf(af, ldsB16(sK, 128, nt * 16, h * 16, lane), sc[nt]);
#pragma unroll
    for (int r = 0; r < 8; ++r) {
      float mx = fmaxf(sc[0][r] * SCALE, sc[1][r] * SCALE);
#pragma unroll
      for (int off = 1; off < 16; off <<= 1) mx = fmaxf(mx, __shfl_xor(mx, off, 32));
      float e0 = __expf(sc[0][r] * SCALE - mx);
      float e1 = __expf(sc[1][r] * SCALE - mx);
      float sum = e0 + e1;
#pragma unroll
      for (int off = 1; off < 16; off <<= 1) sum += __shfl_xor(sum, off, 32);
      float inv = 1.0f / sum;
      Pw[(g * 8 + r) * 32 + n]      = f2bf(e0 * inv);
      Pw[(g * 8 + r) * 32 + 16 + n] = f2bf(e1 * inv);
    }
    v8f o = wmma_bf(ldsA(Pw, 32, 0, 0, lane), ldsB(sVt, 32, h * 16, 0, lane), vzero());
#pragma unroll
    for (int r = 0; r < 8; ++r)
      sO[(q0 + g * 8 + r) * 128 + h * 16 + n] = f2bf(o[r]);
  }

  // output projection + bias -> f32 result
  {
    u32x4* sWv = (u32x4*)sW;
    const u32x4* wv4 = (const u32x4*)wd;
#pragma unroll
    for (int it = 0; it < 8; ++it) sWv[tid + it * 256] = wv4[tid + it * 256];
  }
  __syncthreads();
  v8f acc[2] = {vzero(), vzero()};
#pragma unroll
  for (int kk = 0; kk < 4; ++kk) {
    v16bf a = ldsA(sO, 128, mt * 16, kk * 32, lane);
#pragma unroll
    for (int j = 0; j < 2; ++j)
      acc[j] = wmma_bf(a, ldsB(sW, 128, (ntb + j) * 16, kk * 32, lane), acc[j]);
  }
#pragma unroll
  for (int j = 0; j < 2; ++j) {
#pragma unroll
    for (int r = 0; r < 8; ++r) {
      int tt = mt * 16 + g * 8 + r, d = (ntb + j) * 16 + n;
      out[base + (size_t)tt * DD + d] = acc[j][r] + bd[d];
    }
  }
}

// ---------------------------------------------------------------------------
extern "C" void kernel_launch(void* const* d_in, const int* in_sizes, int n_in,
                              void* d_out, int out_size, void* d_ws, size_t ws_size,
                              hipStream_t stream) {
  (void)in_sizes; (void)n_in; (void)out_size; (void)ws_size;
  const float* q    = (const float*)d_in[0];
  const float* k    = (const float*)d_in[1];
  const float* v    = (const float*)d_in[2];
  const unsigned char* mask = (const unsigned char*)d_in[3];   // bool array
  const float* Wq_t = (const float*)d_in[4];
  const float* Wk_t = (const float*)d_in[5];
  const float* Wv_t = (const float*)d_in[6];
  const float* Wd_t = (const float*)d_in[7];
  const float* bd_t = (const float*)d_in[8];
  const float* Wq_s = (const float*)d_in[9];
  const float* Wk_s = (const float*)d_in[10];
  const float* Wv_s = (const float*)d_in[11];
  const float* Wd_s = (const float*)d_in[12];
  const float* bd_s = (const float*)d_in[13];

  unsigned short* wbf = (unsigned short*)d_ws;        // 8 * 16384 bf16 weights
  unsigned short* xt  = wbf + 8 * 16384;              // [B,L,T,D] bf16 intermediate

  prep_weights_kernel<<<64, 256, 0, stream>>>(
      Wq_t, Wk_t, Wv_t, Wd_t, Wq_s, Wk_s, Wv_s, Wd_s, wbf);

  temporal_kernel<<<BB * TT, 256, 7 * 16384 * sizeof(unsigned short), stream>>>(
      q, k, v, mask,
      wbf + 0 * 16384, wbf + 1 * 16384, wbf + 2 * 16384, wbf + 3 * 16384,
      bd_t, xt);

  social_kernel<<<BB * LL, 256, 40960 * sizeof(unsigned short), stream>>>(
      xt,
      wbf + 4 * 16384, wbf + 5 * 16384, wbf + 6 * 16384, wbf + 7 * 16384,
      bd_s, (float*)d_out);
}
